// VQVAEQuantizer_84705345012107
// MI455X (gfx1250) — matrix operand order repaired and checked by reading, not compiled, not run on hardware
//
#include <hip/hip_runtime.h>

// ---------------------------------------------------------------------------
// VQ-VAE quantizer + decoder for MI455X (gfx1250, wave32, WMMA).
// Block-tiled GEMMs: M=128 x N=128 tile per block. B chunks double-buffered in
// LDS and streamed with CDNA5 async global->LDS loads (ASYNCcnt) so staging
// overlaps WMMA compute. A tile LDS-resident with bank-conflict padding.
// Per wave: 8 accumulators (A-fragment reuse x8) + batched fragment loads so
// the scheduler can issue partial s_wait_dscnt instead of draining per-WMMA.
//  K1: codebook f32 -> bf16 (16MB -> L2-resident)
//  K2: cnorm[k] = |c_k|^2
//  K3/K4: transpose+convert w1,w2 to bf16 column-major for B-fragments
//  K5: fused distance-GEMM + row argmin (never materializes 512MB scores)
//  K6: fused gather(q) + commitment loss + Linear1 + ReLU
//  K7: Linear2 + bias -> recon f32
// ---------------------------------------------------------------------------

typedef __bf16 bf16;
typedef __bf16 v16bf __attribute__((ext_vector_type(16)));
typedef __bf16 v8bf  __attribute__((ext_vector_type(8)));
typedef __bf16 v4bf  __attribute__((ext_vector_type(4)));
typedef float  v8f   __attribute__((ext_vector_type(8)));

#define BT_    16384
#define D_     1024
#define K_     8192
#define H_     512
#define MTILE  128           // rows per block (8 waves x 16-row WMMA tiles)
#define NCHUNK 128           // N columns per block tile iteration
#define DCHUNK 64            // K depth per LDS B stage (double-buffered)
#define APITCH 1032          // 1024 + 8 : stride ≡ 4 dwords mod 64 banks
#define HPITCH 520           // 512 + 8  : stride ≡ 4 dwords mod 64 banks
#define BPITCH 72            // 64 + 8   : stride ≡ 36 dwords mod 64 banks (gcd 4)
#define BBUF_ELEMS (NCHUNK * BPITCH)          // one B buffer

// LDS bytes per block
#define LDS_ARGMIN (MTILE * APITCH * 2 + 2 * BBUF_ELEMS * 2)  // 301,056
#define LDS_MLP2   (MTILE * HPITCH * 2 + 2 * BBUF_ELEMS * 2)  // 169,984

__device__ __forceinline__ bf16 f2bf(float f) { return (bf16)f; }

__device__ __forceinline__ v16bf cat8(v8bf lo, v8bf hi) {
  return __builtin_shufflevector(lo, hi, 0,1,2,3,4,5,6,7,8,9,10,11,12,13,14,15);
}

// A fragment, 16x32 bf16 (ISA 7.12.2): lane half 0 holds K {d0..d0+7, d0+16..23},
// half 1 holds K {d0+8..15, d0+24..31}; M = lane%16 selects the row pointer.
__device__ __forceinline__ v16bf load_a_frag(const bf16* rowPtr, int d0, int half) {
  v8bf lo = *(const v8bf*)(rowPtr + d0 + half * 8);
  v8bf hi = *(const v8bf*)(rowPtr + d0 + 16 + half * 8);
  return cat8(lo, hi);
}

// B fragment, 32x16 bf16: lane column N = lane%16; half 0 holds K {d0..d0+7, d0+16..23},
// half 1 holds K {d0+8..15, d0+24..31} (matches A fragment).
__device__ __forceinline__ v16bf load_b_frag(const bf16* colPtr, int d0, int half) {
  v8bf lo = *(const v8bf*)(colPtr + d0 + half * 8);
  v8bf hi = *(const v8bf*)(colPtr + d0 + 16 + half * 8);
  return cat8(lo, hi);
}

__device__ __forceinline__ v8f wmma_bf16(v16bf a, v16bf b, v8f c) {
  return __builtin_amdgcn_wmma_f32_16x16x32_bf16(false, a, false, b, (short)0, c,
                                                 false, false);
}

// CDNA5 async global -> LDS copy (tracked by ASYNCcnt, no VGPR round-trip).
// Generic shared-aperture pointers keep the LDS byte offset in the low 32 bits.
__device__ __forceinline__ void async_b128(unsigned ldsOff, const void* g) {
  asm volatile("global_load_async_to_lds_b128 %0, %1, off"
               : : "v"(ldsOff), "v"(g) : "memory");
}
__device__ __forceinline__ void wait_async0() {
  asm volatile("s_wait_asynccnt 0x0" : : : "memory");
}

// Issue (no wait) the async stage of a [NCHUNK x DCHUNK] slice of row-major B
// (ldb = K-dim stride) into padded LDS: 1024 x 16B transfers over 256 threads.
__device__ __forceinline__ void stage_b(const bf16* __restrict__ Bg, int ldb,
                                        int n0, int dBase, bf16* ldsB, int tid) {
  #pragma unroll
  for (int k = 0; k < 4; ++k) {
    int i = tid + k * 256;
    int row = i >> 3, c = (i & 7) * 8;
    async_b128((unsigned)(uintptr_t)(ldsB + row * BPITCH + c),
               Bg + (size_t)(n0 + row) * ldb + dBase + c);
  }
}

// One DCHUNK of MMA. Fragments batched ahead of the 8 independent WMMA chains
// so the compiler can emit partial DS waits and overlap LDS latency.
__device__ __forceinline__ void chunk_mma(const bf16* aRowD, const bf16* bColBase,
                                          int half, v8f acc[8]) {
  #pragma unroll
  for (int d0 = 0; d0 < DCHUNK; d0 += 32) {
    v16bf a = load_a_frag(aRowD, d0, half);
    v16bf b[8];
    #pragma unroll
    for (int t = 0; t < 8; ++t)
      b[t] = load_b_frag(bColBase + t * 16 * BPITCH, d0, half);
    #pragma unroll
    for (int t = 0; t < 8; ++t)
      acc[t] = wmma_bf16(a, b[t], acc[t]);
  }
}

// ---- K1: codebook f32 -> bf16 ------------------------------------------------
__global__ void k_convert_cb(const float* __restrict__ cb, bf16* __restrict__ out) {
  const int n4 = (K_ * D_) / 4;
  for (int i = blockIdx.x * blockDim.x + threadIdx.x; i < n4;
       i += gridDim.x * blockDim.x) {
    float4 v = ((const float4*)cb)[i];
    v4bf b; b[0] = f2bf(v.x); b[1] = f2bf(v.y); b[2] = f2bf(v.z); b[3] = f2bf(v.w);
    *(v4bf*)(out + (size_t)i * 4) = b;
  }
}

// ---- K2: cnorm ---------------------------------------------------------------
__global__ void k_cnorm(const float* __restrict__ cb, float* __restrict__ cnorm) {
  const int row = blockIdx.x;
  float4 v = ((const float4*)(cb + (size_t)row * D_))[threadIdx.x]; // 256*4 = 1024
  float s = v.x * v.x + v.y * v.y + v.z * v.z + v.w * v.w;
  for (int m = 16; m >= 1; m >>= 1) s += __shfl_xor(s, m, 32);
  __shared__ float wsum[8];
  if ((threadIdx.x & 31) == 0) wsum[threadIdx.x >> 5] = s;
  __syncthreads();
  if (threadIdx.x == 0) {
    float t = 0.f;
    #pragma unroll
    for (int i = 0; i < 8; ++i) t += wsum[i];
    cnorm[row] = t;
  }
}

// ---- K3: w1 [D,H] -> w1T bf16 [H,D] ; also zero the loss slot ----------------
__global__ void k_prep_w1(const float* __restrict__ w1, bf16* __restrict__ w1T,
                          float* __restrict__ lossSlot) {
  int i = blockIdx.x * blockDim.x + threadIdx.x;
  if (i == 0) *lossSlot = 0.f;
  if (i < D_ * H_) {
    int d = i / H_, h = i % H_;
    w1T[(size_t)h * D_ + d] = f2bf(w1[i]);
  }
}

// ---- K4: w2 [H,D] -> w2T bf16 [D,H] ------------------------------------------
__global__ void k_prep_w2(const float* __restrict__ w2, bf16* __restrict__ w2T) {
  int i = blockIdx.x * blockDim.x + threadIdx.x;
  if (i < H_ * D_) {
    int h = i / D_, d = i % D_;
    w2T[(size_t)d * H_ + h] = f2bf(w2[i]);
  }
}

// ---- K5: fused distance GEMM + argmin ----------------------------------------
__global__ void __launch_bounds__(256)
k_argmin(const float* __restrict__ feat, const bf16* __restrict__ cb,
         const float* __restrict__ cnorm, int* __restrict__ idxInt,
         float* __restrict__ idxOut) {
  extern __shared__ unsigned char smem[];
  bf16* ldsA = (bf16*)smem;                                // [128][APITCH]
  bf16* ldsB0 = (bf16*)(smem + MTILE * APITCH * 2);        // double-buffered B
  bf16* ldsB1 = ldsB0 + BBUF_ELEMS;
  const int tid = threadIdx.x;
  const int rowBase = blockIdx.x * MTILE;

  // stage + convert features tile (padded rows)
  for (int i = tid; i < MTILE * (D_ / 4); i += 256) {
    int row = i / (D_ / 4), c4 = i % (D_ / 4);
    float4 v = ((const float4*)(feat + (size_t)(rowBase + row) * D_))[c4];
    v4bf b; b[0] = f2bf(v.x); b[1] = f2bf(v.y); b[2] = f2bf(v.z); b[3] = f2bf(v.w);
    *(v4bf*)(ldsA + row * APITCH + c4 * 4) = b;
  }

  const int wave = tid >> 5, lane = tid & 31;
  const int half = lane >> 4, lr = lane & 15;
  const bf16* aRow = ldsA + (wave * 16 + lr) * APITCH;

  float bestVal[8];
  int   bestIdx[8];
  #pragma unroll
  for (int r = 0; r < 8; ++r) { bestVal[r] = 3.4e38f; bestIdx[r] = 0; }

  stage_b(cb, D_, 0, 0, ldsB0, tid);             // prefetch first chunk

  for (int n0 = 0; n0 < K_; n0 += NCHUNK) {
    v8f acc[8];
    #pragma unroll
    for (int t = 0; t < 8; ++t)
      acc[t] = (v8f){0.f, 0.f, 0.f, 0.f, 0.f, 0.f, 0.f, 0.f};

    for (int c = 0; c < D_ / DCHUNK; ++c) {      // 16 chunks, even count
      bf16* cur = (c & 1) ? ldsB1 : ldsB0;
      bf16* nxt = (c & 1) ? ldsB0 : ldsB1;
      wait_async0();                             // our stage of `cur` arrived
      __syncthreads();                           // everyone's stage visible;
                                                 // also WAR guard for `nxt`
      int cn = c + 1, nn0 = n0;
      if (cn == D_ / DCHUNK) { cn = 0; nn0 = n0 + NCHUNK; }
      if (nn0 < K_) stage_b(cb, D_, nn0, cn * DCHUNK, nxt, tid);
      chunk_mma(aRow + c * DCHUNK, cur + lr * BPITCH, half, acc);
    }

    #pragma unroll
    for (int t = 0; t < 8; ++t) {
      const int n = n0 + t * 16 + lr;
      const float cn = cnorm[n];
      #pragma unroll
      for (int r = 0; r < 8; ++r) {              // branchless running argmin
        float s = -2.f * acc[t][r] + cn;
        bool lt = s < bestVal[r];
        bestVal[r] = lt ? s : bestVal[r];
        bestIdx[r] = lt ? n : bestIdx[r];
      }
    }
  }

  // min-reduce across the 16 lanes of each half (N dimension), lowest-index ties
  #pragma unroll
  for (int r = 0; r < 8; ++r) {
    float v = bestVal[r]; int ix = bestIdx[r];
    #pragma unroll
    for (int m = 8; m >= 1; m >>= 1) {
      float ov = __shfl_xor(v, m, 32);
      int   oi = __shfl_xor(ix, m, 32);
      bool take = (ov < v) || (ov == v && oi < ix);
      v = take ? ov : v; ix = take ? oi : ix;
    }
    if (lr == 0) {
      int row = rowBase + wave * 16 + half * 8 + r;  // C layout: M = r + 8*half
      idxInt[row] = ix;
      idxOut[row] = (float)ix;
    }
  }
}

// ---- K6: gather q + commitment loss + Linear1 + ReLU -------------------------
__global__ void __launch_bounds__(256)
k_mlp1(const float* __restrict__ feat, const float* __restrict__ cb,
       const int* __restrict__ idxInt, const bf16* __restrict__ w1T,
       const float* __restrict__ b1, bf16* __restrict__ hOut,
       float* __restrict__ lossSlot) {
  extern __shared__ unsigned char smem[];
  bf16* ldsQ = (bf16*)smem;                                // [128][APITCH]
  bf16* ldsB0 = (bf16*)(smem + MTILE * APITCH * 2);
  bf16* ldsB1 = ldsB0 + BBUF_ELEMS;
  const int tid = threadIdx.x;
  const int rowBase = blockIdx.x * MTILE;

  float lsum = 0.f;
  for (int i = tid; i < MTILE * (D_ / 4); i += 256) {
    int row = i / (D_ / 4), c4 = i % (D_ / 4);
    int idx = idxInt[rowBase + row];
    float4 q = ((const float4*)(cb + (size_t)idx * D_))[c4];
    float4 x = ((const float4*)(feat + (size_t)(rowBase + row) * D_))[c4];
    float d0 = x.x - q.x, d1 = x.y - q.y, d2 = x.z - q.z, d3 = x.w - q.w;
    lsum += d0 * d0 + d1 * d1 + d2 * d2 + d3 * d3;
    v4bf b; b[0] = f2bf(q.x); b[1] = f2bf(q.y); b[2] = f2bf(q.z); b[3] = f2bf(q.w);
    *(v4bf*)(ldsQ + row * APITCH + c4 * 4) = b;            // STE forward == q
  }
  for (int m = 16; m >= 1; m >>= 1) lsum += __shfl_xor(lsum, m, 32);
  if ((tid & 31) == 0)
    atomicAdd(lossSlot, lsum * (0.25f / ((float)BT_ * (float)D_)));

  const int wave = tid >> 5, lane = tid & 31;
  const int half = lane >> 4, lr = lane & 15;
  const bf16* aRow = ldsQ + (wave * 16 + lr) * APITCH;

  stage_b(w1T, D_, 0, 0, ldsB0, tid);

  for (int n0 = 0; n0 < H_; n0 += NCHUNK) {
    v8f acc[8];
    #pragma unroll
    for (int t = 0; t < 8; ++t)
      acc[t] = (v8f){0.f, 0.f, 0.f, 0.f, 0.f, 0.f, 0.f, 0.f};

    for (int c = 0; c < D_ / DCHUNK; ++c) {
      bf16* cur = (c & 1) ? ldsB1 : ldsB0;
      bf16* nxt = (c & 1) ? ldsB0 : ldsB1;
      wait_async0();
      __syncthreads();
      int cn = c + 1, nn0 = n0;
      if (cn == D_ / DCHUNK) { cn = 0; nn0 = n0 + NCHUNK; }
      if (nn0 < H_) stage_b(w1T, D_, nn0, cn * DCHUNK, nxt, tid);
      chunk_mma(aRow + c * DCHUNK, cur + lr * BPITCH, half, acc);
    }

    #pragma unroll
    for (int t = 0; t < 8; ++t) {
      const int col = n0 + t * 16 + lr;
      const float bias = b1[col];
      #pragma unroll
      for (int r = 0; r < 8; ++r) {
        float v = acc[t][r] + bias;
        v = v > 0.f ? v : 0.f;                   // ReLU
        int row = rowBase + wave * 16 + half * 8 + r;
        hOut[(size_t)row * H_ + col] = f2bf(v);
      }
    }
  }
}

// ---- K7: Linear2 + bias -> recon (f32) ---------------------------------------
__global__ void __launch_bounds__(256)
k_mlp2(const bf16* __restrict__ hIn, const bf16* __restrict__ w2T,
       const float* __restrict__ b2, float* __restrict__ recon) {
  extern __shared__ unsigned char smem[];
  bf16* ldsH = (bf16*)smem;                                // [128][HPITCH]
  bf16* ldsB0 = (bf16*)(smem + MTILE * HPITCH * 2);
  bf16* ldsB1 = ldsB0 + BBUF_ELEMS;
  const int tid = threadIdx.x;
  const int rowBase = blockIdx.x * MTILE;

  // async-stage the h tile (pure bf16 copy, padded rows)
  for (int i = tid; i < MTILE * (H_ / 8); i += 256) {
    int row = i >> 6, c = (i & 63) * 8;
    async_b128((unsigned)(uintptr_t)(ldsH + row * HPITCH + c),
               hIn + (size_t)(rowBase + row) * H_ + c);
  }

  const int wave = tid >> 5, lane = tid & 31;
  const int half = lane >> 4, lr = lane & 15;
  const bf16* aRow = ldsH + (wave * 16 + lr) * HPITCH;

  stage_b(w2T, H_, 0, 0, ldsB0, tid);

  for (int n0 = 0; n0 < D_; n0 += NCHUNK) {
    v8f acc[8];
    #pragma unroll
    for (int t = 0; t < 8; ++t)
      acc[t] = (v8f){0.f, 0.f, 0.f, 0.f, 0.f, 0.f, 0.f, 0.f};

    for (int c = 0; c < H_ / DCHUNK; ++c) {      // 8 chunks, even count
      bf16* cur = (c & 1) ? ldsB1 : ldsB0;
      bf16* nxt = (c & 1) ? ldsB0 : ldsB1;
      wait_async0();                             // covers h-tile + B stage
      __syncthreads();
      int cn = c + 1, nn0 = n0;
      if (cn == H_ / DCHUNK) { cn = 0; nn0 = n0 + NCHUNK; }
      if (nn0 < D_) stage_b(w2T, H_, nn0, cn * DCHUNK, nxt, tid);
      chunk_mma(aRow + c * DCHUNK, cur + lr * BPITCH, half, acc);
    }

    #pragma unroll
    for (int t = 0; t < 8; ++t) {
      const int col = n0 + t * 16 + lr;
      const float bias = b2[col];
      #pragma unroll
      for (int r = 0; r < 8; ++r) {
        int row = rowBase + wave * 16 + half * 8 + r;
        recon[(size_t)row * D_ + col] = acc[t][r] + bias;
      }
    }
  }
}

// ---------------------------------------------------------------------------
extern "C" void kernel_launch(void* const* d_in, const int* in_sizes, int n_in,
                              void* d_out, int out_size, void* d_ws, size_t ws_size,
                              hipStream_t stream) {
  const float* feat = (const float*)d_in[0];   // [8,2048,1024]
  const float* cb   = (const float*)d_in[1];   // [8192,1024]
  const float* w1   = (const float*)d_in[2];   // [1024,512]
  const float* b1   = (const float*)d_in[3];   // [512]
  const float* w2   = (const float*)d_in[4];   // [512,1024]
  const float* b2   = (const float*)d_in[5];   // [1024]

  float* out = (float*)d_out;
  float* recon   = out;                        // 16,777,216 f32
  float* idxOut  = out + (size_t)BT_ * D_;     // 16,384
  float* lossPtr = idxOut + BT_;               // 1

  // workspace carve-up
  unsigned char* ws = (unsigned char*)d_ws;
  bf16*  cb_bf  = (bf16*)ws;                              ws += (size_t)K_ * D_ * 2;
  float* cnorm  = (float*)ws;                             ws += (size_t)K_ * 4;
  bf16*  w1T    = (bf16*)ws;                              ws += (size_t)H_ * D_ * 2;
  bf16*  w2T    = (bf16*)ws;                              ws += (size_t)D_ * H_ * 2;
  int*   idxInt = (int*)ws;                               ws += (size_t)BT_ * 4;
  bf16*  hBuf   = (bf16*)ws;                              // [BT, H] bf16

  k_convert_cb<<<2048, 256, 0, stream>>>(cb, cb_bf);
  k_cnorm<<<K_, 256, 0, stream>>>(cb, cnorm);
  k_prep_w1<<<(D_ * H_ + 255) / 256, 256, 0, stream>>>(w1, w1T, lossPtr);
  k_prep_w2<<<(H_ * D_ + 255) / 256, 256, 0, stream>>>(w2, w2T);

  const int blocks = BT_ / MTILE;              // 128
  k_argmin<<<blocks, 256, LDS_ARGMIN, stream>>>(feat, cb_bf, cnorm, idxInt, idxOut);
  k_mlp1<<<blocks, 256, LDS_ARGMIN, stream>>>(feat, cb, idxInt, w1T, b1, hBuf, lossPtr);
  k_mlp2<<<blocks, 256, LDS_MLP2, stream>>>(hBuf, w2T, b2, recon);
}
